// RecurrentNeuronalNetwork_67259187855681
// MI455X (gfx1250) — matrix operand order repaired and checked by reading, not compiled, not run on hardware
//
#include <hip/hip_runtime.h>
#include <hip/hip_bf16.h>
#include <math.h>

// ---- problem dims (fixed by the reference) ----
#define D_IN  512
#define HID   1024
#define D_OUT 32
#define NB    128     // batch
#define NT    512     // seq len
#define MTOT  (NB*NT) // 65536 rows for the input-projection GEMM
#define NBLK_SCAN 32  // persistent-scan workgroups (2 x 16)

typedef __bf16 bf16;
typedef __attribute__((ext_vector_type(16))) __bf16 v16bf;
typedef __attribute__((ext_vector_type(8)))  __bf16 bf16x8;
typedef __attribute__((ext_vector_type(4)))  __bf16 bf16x4;
typedef __attribute__((ext_vector_type(8)))  float  v8f;
typedef __attribute__((ext_vector_type(4)))  float  f32x4;

// ---------------------------------------------------------------------------
// WMMA fragment layouts used by this code (wave32; A/B k order is contiguous, not the ISA table order):
//  A (16x32 bf16): lane%16 = M row; lanes 0-15 hold K=0..15,
//                  lanes 16-31 hold K=16..31  -> two contiguous 16B runs.
//  B (32x16 bf16): lane%16 = N col; lanes 0-15 hold K=0..15, 16-31 K=16..31.
//                  B tiles stored transposed (Bs[n][k]) -> one 32B run per lane.
//  C/D (16x16 f32): VGPR r -> M = r + 8*(lane/16), N = lane%16.
// ---------------------------------------------------------------------------
__device__ __forceinline__ v16bf load_frag_a_lds(const bf16* As, int m0, int lane) {
    int m = m0 + (lane & 15);
    int half = lane >> 4;
    bf16x8 lo = *(const bf16x8*)(As + m * 32 + half * 16);
    bf16x8 hi = *(const bf16x8*)(As + m * 32 + half * 16 + 8);
    v16bf a;
#pragma unroll
    for (int i = 0; i < 8; ++i) { a[i] = lo[i]; a[i + 8] = hi[i]; }
    return a;
}

__device__ __forceinline__ v16bf load_frag_b_lds(const bf16* Bs, int n0, int lane) {
    int n = n0 + (lane & 15);
    int half = lane >> 4;
    bf16x8 lo = *(const bf16x8*)(Bs + n * 32 + half * 16);
    bf16x8 hi = *(const bf16x8*)(Bs + n * 32 + half * 16 + 8);
    v16bf b;
#pragma unroll
    for (int i = 0; i < 8; ++i) { b[i] = lo[i]; b[i + 8] = hi[i]; }
    return b;
}

__device__ __forceinline__ v8f wmma_bf16(v16bf a, v16bf b, v8f c) {
    return __builtin_amdgcn_wmma_f32_16x16x32_bf16(false, a, false, b, (short)0, c,
                                                   false, false);
}

// ---------------------------------------------------------------------------
// Init: zero h0 (128x1024 bf16) and the grid-barrier counter.
// grid 256 x 256 threads, uint2 per thread covers 512 KB (hbuf0 + hbuf1).
// ---------------------------------------------------------------------------
__global__ __launch_bounds__(256) void init_state(bf16* __restrict__ h,
                                                  unsigned* __restrict__ cnt) {
    int idx = blockIdx.x * 256 + threadIdx.x;
    ((uint2*)h)[idx] = make_uint2(0u, 0u);
    if (idx == 0) *cnt = 0u;
}

// ---------------------------------------------------------------------------
// GEMM1: xproj[M=65536, N=1024] = x[M,512] @ W_xh[512,1024] + b_h
// f32 inputs converted to bf16 while staging into LDS. 64x64 block / 4 waves.
// ---------------------------------------------------------------------------
__global__ __launch_bounds__(128) void gemm_xproj(const float* __restrict__ x,
                                                  const float* __restrict__ Wxh,
                                                  const float* __restrict__ bh,
                                                  float* __restrict__ xproj) {
    __shared__ __align__(16) bf16 As[64 * 32];
    __shared__ __align__(16) bf16 Bs[64 * 32];
    const int tid = threadIdx.x, lane = tid & 31, wave = tid >> 5;
    const int wm = wave & 1, wn = wave >> 1;
    const int m_blk = blockIdx.x * 64;
    const int n_blk = blockIdx.y * 64;

    v8f c[2][2] = {};
    for (int k0 = 0; k0 < D_IN; k0 += 32) {
        // stage A: 64x32 f32 tile of x -> bf16 LDS (row-major)
#pragma unroll
        for (int i = 0; i < 4; ++i) {
            int idx = tid + i * 128;            // 512 float4 chunks
            int row = idx >> 3, col = (idx & 7) << 2;
            f32x4 v = *(const f32x4*)(x + (size_t)(m_blk + row) * D_IN + k0 + col);
            bf16x4 w; w[0] = (bf16)v[0]; w[1] = (bf16)v[1]; w[2] = (bf16)v[2]; w[3] = (bf16)v[3];
            *(bf16x4*)(As + row * 32 + col) = w;
        }
        // stage B: 32x64 tile of W_xh -> bf16 LDS transposed (Bs[n][k])
#pragma unroll
        for (int i = 0; i < 4; ++i) {
            int idx = tid + i * 128;
            int row = idx >> 4, col = (idx & 15) << 2;
            f32x4 v = *(const f32x4*)(Wxh + (size_t)(k0 + row) * HID + n_blk + col);
#pragma unroll
            for (int j = 0; j < 4; ++j) Bs[(col + j) * 32 + row] = (bf16)v[j];
        }
        __syncthreads();
        if (k0 + 32 < D_IN)
            __builtin_prefetch(x + (size_t)(m_blk + (tid >> 3)) * D_IN + k0 + 32, 0, 1);

        v16bf bfr[2];
        bfr[0] = load_frag_b_lds(Bs, wn * 32, lane);
        bfr[1] = load_frag_b_lds(Bs, wn * 32 + 16, lane);
#pragma unroll
        for (int i = 0; i < 2; ++i) {
            v16bf a = load_frag_a_lds(As, wm * 32 + i * 16, lane);
#pragma unroll
            for (int j = 0; j < 2; ++j) c[i][j] = wmma_bf16(a, bfr[j], c[i][j]);
        }
        __syncthreads();
    }
    // epilogue: + b_h, f32 store
    const int nc = lane & 15, half = lane >> 4;
#pragma unroll
    for (int i = 0; i < 2; ++i)
#pragma unroll
        for (int j = 0; j < 2; ++j) {
            int col = n_blk + wn * 32 + j * 16 + nc;
            float bias = bh[col];
#pragma unroll
            for (int r = 0; r < 8; ++r) {
                int row = m_blk + wm * 32 + i * 16 + r + 8 * half;
                xproj[(size_t)row * HID + col] = c[i][j][r] + bias;
            }
        }
}

// ---------------------------------------------------------------------------
// Persistent recurrent scan: ONE launch runs all 512 timesteps.
//   h_{t+1} = tanh(xproj[:,t,:] + h_t @ W_hh)
// Grid = (2, 16) = 32 blocks of 128 threads. Each block owns a 1024x64 column
// panel of W_hh, converted f32->bf16 once into a 128 KB dynamic-LDS panel
// (CDNA5: 320 KB/WGP), stored transposed so B-fragments are contiguous 32B
// ds loads. Steady-state K-loop has NO __syncthreads: LDS is read-only,
// A-fragments stream straight from the L2-resident h ping-pong buffers.
// Steps are separated by a software device-wide barrier (32 co-resident
// blocks; atomic counter + s_sleep spin, release/acquire fences).
// ---------------------------------------------------------------------------
__global__ __launch_bounds__(128) void rnn_scan(bf16* __restrict__ hbuf0,
                                                bf16* __restrict__ hbuf1,
                                                const float* __restrict__ Whh,
                                                const float* __restrict__ xproj,
                                                unsigned* __restrict__ cnt) {
    extern __shared__ bf16 Bs[];  // [64][1024] transposed W_hh panel (128 KB)
    const int tid = threadIdx.x, lane = tid & 31, wave = tid >> 5;
    const int wm = wave & 1, wn = wave >> 1;
    const int m_blk = blockIdx.x * 64;
    const int n_blk = blockIdx.y * 64;
    const int half = lane >> 4;

    // one-time: stage W_hh[:, n_blk:n_blk+64] f32 -> bf16, transposed
    for (int i = tid; i < (HID * 64) / 4; i += 128) {   // 16384 float4 chunks
        int k = i >> 4;                 // 0..1023
        int c4 = (i & 15) << 2;         // 0..60
        f32x4 v = *(const f32x4*)(Whh + (size_t)k * HID + n_blk + c4);
#pragma unroll
        for (int j = 0; j < 4; ++j) Bs[(size_t)(c4 + j) * HID + k] = (bf16)v[j];
    }
    __syncthreads();

    bf16* hb[2] = { hbuf0, hbuf1 };
    for (int t = 0; t < NT; ++t) {
        const bf16* h_in = hb[t & 1];
        bf16* h_out = hb[(t + 1) & 1];
        const float* xp_t = xproj + (size_t)t * HID;

        v8f c[2][2] = {};
        for (int k0 = 0; k0 < HID; k0 += 32) {
            // B fragments from the resident LDS panel (stride HID)
            v16bf bfr[2];
#pragma unroll
            for (int j = 0; j < 2; ++j) {
                const bf16* bp = Bs + (size_t)(wn * 32 + j * 16 + (lane & 15)) * HID
                                 + k0 + half * 16;
                bf16x8 lo = *(const bf16x8*)bp;
                bf16x8 hi = *(const bf16x8*)(bp + 8);
#pragma unroll
                for (int e = 0; e < 8; ++e) { bfr[j][e] = lo[e]; bfr[j][e + 8] = hi[e]; }
            }
            // A fragments straight from global h_in (L2-resident, 2x16B/lane)
#pragma unroll
            for (int i = 0; i < 2; ++i) {
                const bf16* ap = h_in
                    + (size_t)(m_blk + wm * 32 + i * 16 + (lane & 15)) * HID
                    + k0 + half * 16;
                bf16x8 lo = *(const bf16x8*)ap;
                bf16x8 hi = *(const bf16x8*)(ap + 8);
                v16bf a;
#pragma unroll
                for (int e = 0; e < 8; ++e) { a[e] = lo[e]; a[e + 8] = hi[e]; }
#pragma unroll
                for (int j = 0; j < 2; ++j) c[i][j] = wmma_bf16(a, bfr[j], c[i][j]);
            }
        }
        // epilogue: + xproj_t, tanh, bf16 store into the other ping-pong buffer
        const int nc = lane & 15;
#pragma unroll
        for (int i = 0; i < 2; ++i)
#pragma unroll
            for (int j = 0; j < 2; ++j) {
                int col = n_blk + wn * 32 + j * 16 + nc;
#pragma unroll
                for (int r = 0; r < 8; ++r) {
                    int row = m_blk + wm * 32 + i * 16 + r + 8 * half;  // batch idx
                    float v = c[i][j][r] + xp_t[(size_t)row * (NT * HID) + col];
                    h_out[(size_t)row * HID + col] = (bf16)tanhf(v);
                }
            }

        // ---- device-wide barrier between timesteps ----
        __syncthreads();
        __threadfence();                       // release: h_out visible device-wide
        if (tid == 0) {
            __hip_atomic_fetch_add(cnt, 1u, __ATOMIC_RELEASE,
                                   __HIP_MEMORY_SCOPE_AGENT);
            const unsigned target = (unsigned)NBLK_SCAN * (unsigned)(t + 1);
            while (__hip_atomic_load(cnt, __ATOMIC_ACQUIRE,
                                     __HIP_MEMORY_SCOPE_AGENT) < target)
                __builtin_amdgcn_s_sleep(1);
        }
        __syncthreads();
        __threadfence();                       // acquire: invalidate before reads
    }
}

// ---------------------------------------------------------------------------
// Final FC: out[128,32] = h_T @ fc_w^T + fc_b.  One 16-row tile per block,
// 2 waves = 2 N-tiles of 16.  Fragments loaded straight from global.
// ---------------------------------------------------------------------------
__global__ __launch_bounds__(64) void fc_head(const bf16* __restrict__ h,
                                              const float* __restrict__ fcw,
                                              const float* __restrict__ fcb,
                                              float* __restrict__ out) {
    const int lane = threadIdx.x & 31, wave = threadIdx.x >> 5;
    const int m0 = blockIdx.x * 16;
    const int n0 = wave * 16;
    const int m = lane & 15, half = lane >> 4;
    v8f c = {};
    for (int k0 = 0; k0 < HID; k0 += 32) {
        const bf16* ap = h + (size_t)(m0 + m) * HID + k0;
        bf16x8 lo = *(const bf16x8*)(ap + half * 16);
        bf16x8 hi = *(const bf16x8*)(ap + half * 16 + 8);
        v16bf a;
#pragma unroll
        for (int i = 0; i < 8; ++i) { a[i] = lo[i]; a[i + 8] = hi[i]; }
        // B[k][n] = fc_w[n][k] : 16 contiguous f32 per lane, convert to bf16
        const float* bp = fcw + (size_t)(n0 + (lane & 15)) * HID + k0 + half * 16;
        v16bf b;
#pragma unroll
        for (int q = 0; q < 4; ++q) {
            f32x4 v = *(const f32x4*)(bp + q * 4);
#pragma unroll
            for (int j = 0; j < 4; ++j) b[q * 4 + j] = (bf16)v[j];
        }
        c = wmma_bf16(a, b, c);
    }
    const int n = n0 + (lane & 15);
    float bias = fcb[n];
#pragma unroll
    for (int r = 0; r < 8; ++r)
        out[(size_t)(m0 + r + 8 * half) * D_OUT + n] = c[r] + bias;
}

// ---------------------------------------------------------------------------
extern "C" void kernel_launch(void* const* d_in, const int* in_sizes, int n_in,
                              void* d_out, int out_size, void* d_ws, size_t ws_size,
                              hipStream_t stream) {
    (void)in_sizes; (void)n_in; (void)out_size; (void)ws_size;
    const float* x   = (const float*)d_in[0];  // (128, 512, 512)
    const float* Wxh = (const float*)d_in[1];  // (512, 1024)
    const float* Whh = (const float*)d_in[2];  // (1024, 1024)
    const float* bh  = (const float*)d_in[3];  // (1024,)
    const float* fcw = (const float*)d_in[4];  // (32, 1024)
    const float* fcb = (const float*)d_in[5];  // (32,)
    float* out = (float*)d_out;                // (128, 32)

    char* ws = (char*)d_ws;
    float* xproj   = (float*)ws;                                  // 256 MB f32
    size_t off     = (size_t)MTOT * HID * sizeof(float);
    bf16* hbuf0    = (bf16*)(ws + off);                           // 128x1024 bf16
    bf16* hbuf1    = hbuf0 + (size_t)NB * HID;
    off           += 2 * (size_t)NB * HID * sizeof(bf16);
    unsigned* cnt  = (unsigned*)(ws + off);                       // barrier counter

    init_state<<<dim3((NB * HID / 2) / 256), 256, 0, stream>>>(hbuf0, cnt);

    gemm_xproj<<<dim3(MTOT / 64, HID / 64), 128, 0, stream>>>(x, Wxh, bh, xproj);

    // one persistent launch for all 512 timesteps; 128 KB dynamic LDS panel
    rnn_scan<<<dim3(NB / 64, HID / 64), 128, 64 * HID * sizeof(bf16), stream>>>(
        hbuf0, hbuf1, Whh, xproj, cnt);

    // NT = 512 is even -> final hidden state is in hbuf0
    fc_head<<<dim3(NB / 16), 64, 0, stream>>>(hbuf0, fcw, fcb, out);
}
